// SpikeMLP_69415261438436
// MI455X (gfx1250) — matrix operand order repaired and checked by reading, not compiled, not run on hardware
//
#include <hip/hip_runtime.h>
#include <hip/hip_bf16.h>
#include <math.h>

typedef __attribute__((ext_vector_type(16))) _Float16 v16h;
typedef __attribute__((ext_vector_type(8)))  _Float16 v8h;
typedef __attribute__((ext_vector_type(8)))  float    v8f;

#define B_SZ    1024
#define D_IN    4096
#define HID     256
#define D_OUT   80
#define T_STEPS 32

// gfx1250 async memory->LDS DMA (per-lane 16B), tracked with ASYNCcnt.
#define GL_ASYNC_B128(ldsaddr, gaddr)                            \
  asm volatile("global_load_async_to_lds_b128 %0, %1, off"       \
               :: "v"(ldsaddr), "v"(gaddr) : "memory")
#define WAIT_ASYNC0() asm volatile("s_wait_asynccnt 0x0" ::: "memory")

// ---------- ordered-float encoding so atomicMax/Min on u32 order like floats ----------
__device__ __forceinline__ unsigned encF(float f) {
  unsigned u = __float_as_uint(f);
  return (u & 0x80000000u) ? ~u : (u | 0x80000000u);
}
__device__ __forceinline__ float decF(unsigned u) {
  u = (u & 0x80000000u) ? (u & 0x7fffffffu) : ~u;
  return __uint_as_float(u);
}

// LDS byte offset of a __shared__ pointer: low 32 bits of the flat address
__device__ __forceinline__ unsigned lds_off(const void* p) {
  return (unsigned)(size_t)p;
}

// ---------- f32 -> f16 convert (weights) ----------
__global__ void f32_to_f16(const float* __restrict__ in, _Float16* __restrict__ out, int n) {
  int i = blockIdx.x * 256 + threadIdx.x;
  if (i < n) out[i] = (_Float16)in[i];
}

// ---------- stats slots: [sum_v, max_v(enc), min_v(enc), sum_t, max_t(enc), min_t(enc), pad, pad] ----------
__global__ void init_stats(unsigned* __restrict__ s, int n) {
  int i = blockIdx.x * 256 + threadIdx.x;
  if (i < n) {
    int f = i & 7;
    s[i] = (f == 2 || f == 5) ? 0xFFFFFFFFu : 0u;  // min slots -> +inf encoding floor
  }
}

// ---------- spikes (B, D_IN, T) f32  ->  Xh (T, B, D_IN) f16, LDS tile transpose ----------
__global__ void transpose_spikes(const float* __restrict__ S, _Float16* __restrict__ Xh) {
  __shared__ float tile[32][33];
  const int b  = blockIdx.y;
  const int k0 = blockIdx.x * 32;
  const int tx = threadIdx.x;            // t index on load, k index on store
  for (int i = threadIdx.y; i < 32; i += 8)
    tile[i][tx] = S[((size_t)b * D_IN + k0 + i) * T_STEPS + tx];   // 128B coalesced along t
  __syncthreads();
  for (int i = threadIdx.y; i < 32; i += 8)                         // i = t, tx = k
    Xh[((size_t)i * B_SZ + b) * D_IN + k0 + tx] = (_Float16)tile[tx][i];
}

// ---------- tiled WMMA GEMM: C[M,N] = A[M,K]f16 * W[N,K]f16^T + bias[N] ----------
// 128 threads (4 waves); block tile 64x64; wave tile 16x64; K step 32.
// Double-buffered LDS filled by global_load_async_to_lds_b128 (ASYNCcnt).
// M % 64 == 0, K % 32 == 0; N guarded via pre-zeroed LDS rows (N=80 case).
__launch_bounds__(128)
__global__ void gemm_f16_wmma(const _Float16* __restrict__ A,
                              const _Float16* __restrict__ W,
                              const float* __restrict__ bias,
                              float* __restrict__ C,
                              int M, int N, int K) {
  __shared__ __align__(16) _Float16 As[2][64 * 40];   // padded rows: 40 halfs
  __shared__ __align__(16) _Float16 Bs[2][64 * 40];

  const int tid  = threadIdx.x;
  const int wave = tid >> 5;
  const int lane = tid & 31;
  const int row  = lane & 15;
  const int hs   = lane >> 4;            // half-wave select
  const int m0   = blockIdx.y * 64;
  const int n0   = blockIdx.x * 64;

  // each thread owns two fixed 16B chunks of each 64x32 tile
  const int r0 = tid >> 2;               // rows 0..31
  const int r1 = r0 + 32;                // rows 32..63
  const int pp = (tid & 3) * 8;          // half offset within row

  const _Float16* gA0 = A + (size_t)(m0 + r0) * K + pp;
  const _Float16* gA1 = A + (size_t)(m0 + r1) * K + pp;
  const _Float16* gB0 = W + (size_t)(n0 + r0) * K + pp;
  const _Float16* gB1 = W + (size_t)(n0 + r1) * K + pp;
  const bool okB0 = (n0 + r0) < N;
  const bool okB1 = (n0 + r1) < N;

  unsigned lA0[2], lA1[2], lB0[2], lB1[2];
#pragma unroll
  for (int b = 0; b < 2; ++b) {
    lA0[b] = lds_off(&As[b][r0 * 40 + pp]);
    lA1[b] = lds_off(&As[b][r1 * 40 + pp]);
    lB0[b] = lds_off(&Bs[b][r0 * 40 + pp]);
    lB1[b] = lds_off(&Bs[b][r1 * 40 + pp]);
  }
  // out-of-range W rows: zero once, never overwritten by async loads
  if (!okB0) { *(v8h*)&Bs[0][r0 * 40 + pp] = (v8h){}; *(v8h*)&Bs[1][r0 * 40 + pp] = (v8h){}; }
  if (!okB1) { *(v8h*)&Bs[0][r1 * 40 + pp] = (v8h){}; *(v8h*)&Bs[1][r1 * 40 + pp] = (v8h){}; }

  v8f acc[4] = {};

  auto issue = [&](int nb, int k0) {
    GL_ASYNC_B128(lA0[nb], (unsigned long long)(size_t)(gA0 + k0));
    GL_ASYNC_B128(lA1[nb], (unsigned long long)(size_t)(gA1 + k0));
    if (okB0) GL_ASYNC_B128(lB0[nb], (unsigned long long)(size_t)(gB0 + k0));
    if (okB1) GL_ASYNC_B128(lB1[nb], (unsigned long long)(size_t)(gB1 + k0));
  };

  issue(0, 0);
  int buf = 0;
  for (int k0 = 0; k0 < K; k0 += 32) {
    WAIT_ASYNC0();          // this wave's async loads into buf are done
    __syncthreads();        // everyone's are done; prior readers of buf^1 are done
    if (k0 + 32 < K) issue(buf ^ 1, k0 + 32);   // prefetch next tile

    const _Float16* as = &As[buf][0];
    const _Float16* bs = &Bs[buf][0];

    // A fragment (16x32 f16): lane row, K = hs*8 + i&7  and  16 + hs*8 + i&7
    union F16x16 { v16h v; v8h h[2]; };
    F16x16 af;
    const int ar = wave * 16 + row;
    af.h[0] = *(const v8h*)&as[ar * 40 + hs * 8];
    af.h[1] = *(const v8h*)&as[ar * 40 + 16 + hs * 8];

#pragma unroll
    for (int j = 0; j < 4; ++j) {
      // B fragment (32x16 f16): lane col, K = hs*8 + i&7  and  16 + hs*8 + i&7
      F16x16 bf;
      const int bc = j * 16 + row;
      bf.h[0] = *(const v8h*)&bs[bc * 40 + hs * 8];
      bf.h[1] = *(const v8h*)&bs[bc * 40 + 16 + hs * 8];
      acc[j] = __builtin_amdgcn_wmma_f32_16x16x32_f16(
          false, af.v, false, bf.v, (short)0, acc[j], false, false);
    }
    buf ^= 1;
  }

  // C/D layout: VGPR r -> M = r + hs*8, N = lane&15
#pragma unroll
  for (int j = 0; j < 4; ++j) {
    const int n = n0 + j * 16 + row;
    if (n < N) {
      const float bv = bias[n];
#pragma unroll
      for (int r8 = 0; r8 < 8; ++r8) {
        const int m = m0 + wave * 16 + hs * 8 + r8;
        C[(size_t)m * N + n] = acc[j][r8] + bv;
      }
    }
  }
}

// ---------- LIF + dynamic threshold + temporal(prev stats) + stats(cur) ----------
__launch_bounds__(256)
__global__ void lif_step(const float* __restrict__ Z,       // GEMM result incl. bias
                         float* __restrict__ c, float* __restrict__ v,
                         _Float16* __restrict__ sh,         // spike state (0/1) as f16
                         const unsigned* __restrict__ statsPrev,
                         unsigned* __restrict__ statsCur,
                         float* __restrict__ act,           // nullable (layer 3 only)
                         int n, int step) {
  const int i = blockIdx.x * 256 + threadIdx.x;

  float vm = 0.f, vthbar = 0.f;
  if (step != 0) {
    const float sum_v = __uint_as_float(statsPrev[0]);
    const float max_v = decF(statsPrev[1]);
    const float min_v = decF(statsPrev[2]);
    const float sum_t = __uint_as_float(statsPrev[3]);
    const float max_t = decF(statsPrev[4]);
    const float min_t = decF(statsPrev[5]);
    vm     = sum_v / (float)n - 0.2f * (max_v - min_v);
    vthbar = sum_t / (float)n - 0.2f * (max_t - min_t);
  }

  float vn = 0.f, vthn = 0.f;
  const bool valid = (i < n);
  if (valid) {
    const float vprev = v[i];
    const float tpr = (step == 0)
        ? 0.5f
        : (0.01f * (vprev - vm) + vthbar + log1pf(expf((vprev - vm) * (1.f / 6.f))));
    const float sprev = (float)sh[i];
    const float cn = c[i] * 0.5f + Z[i];
    vn = vprev * 0.75f * (1.f - sprev) + cn;
    const float energy = expf((vprev - vn) * (1.f / 3.f)) - 1.f;
    vthn = 0.5f * tpr + 0.5f * energy;
    const float sn = (vn > vthn) ? 1.f : 0.f;
    c[i]  = cn;
    v[i]  = vn;
    sh[i] = (_Float16)sn;
    if (act) act[i] += sn;
  }

  // wave32 reduction, then one set of atomics per wave
  float rsv = valid ? vn : 0.f,          rst = valid ? vthn : 0.f;
  float rxv = valid ? vn : -3.4e38f,     rnv = valid ? vn : 3.4e38f;
  float rxt = valid ? vthn : -3.4e38f,   rnt = valid ? vthn : 3.4e38f;
#pragma unroll
  for (int off = 16; off; off >>= 1) {
    rsv += __shfl_xor(rsv, off, 32);
    rst += __shfl_xor(rst, off, 32);
    rxv = fmaxf(rxv, __shfl_xor(rxv, off, 32));
    rnv = fminf(rnv, __shfl_xor(rnv, off, 32));
    rxt = fmaxf(rxt, __shfl_xor(rxt, off, 32));
    rnt = fminf(rnt, __shfl_xor(rnt, off, 32));
  }
  if ((threadIdx.x & 31) == 0) {
    atomicAdd((float*)&statsCur[0], rsv);
    atomicMax(&statsCur[1], encF(rxv));
    atomicMin(&statsCur[2], encF(rnv));
    atomicAdd((float*)&statsCur[3], rst);
    atomicMax(&statsCur[4], encF(rxt));
    atomicMin(&statsCur[5], encF(rnt));
  }
}

__global__ void finalize_out(const float* __restrict__ act, float* __restrict__ out, int n) {
  int i = blockIdx.x * 256 + threadIdx.x;
  if (i < n) out[i] = act[i] * (1.0f / (float)T_STEPS);
}

extern "C" void kernel_launch(void* const* d_in, const int* in_sizes, int n_in,
                              void* d_out, int out_size, void* d_ws, size_t ws_size,
                              hipStream_t stream) {
  (void)in_sizes; (void)n_in; (void)out_size; (void)ws_size;
  const float* spikes = (const float*)d_in[0];
  const float* W1 = (const float*)d_in[1];
  const float* b1 = (const float*)d_in[2];
  const float* W2 = (const float*)d_in[3];
  const float* b2 = (const float*)d_in[4];
  const float* W3 = (const float*)d_in[5];
  const float* b3 = (const float*)d_in[6];

  // ---- workspace carve-up (~312 MB total; all re-initialized every call) ----
  char* ws = (char*)d_ws;
  size_t off = 0;
  auto take = [&](size_t bytes) -> char* {
    char* p = ws + off;
    off = (off + bytes + 255) & ~(size_t)255;
    return p;
  };
  _Float16* Xh  = (_Float16*)take((size_t)T_STEPS * B_SZ * D_IN * 2);   // 256 MiB
  float*    Z1  = (float*)take((size_t)T_STEPS * B_SZ * HID * 4);       // 32 MiB
  _Float16* W1h = (_Float16*)take((size_t)HID * D_IN * 2);
  _Float16* W2h = (_Float16*)take((size_t)HID * HID * 2);
  _Float16* W3h = (_Float16*)take((size_t)D_OUT * HID * 2);
  float*    Z2  = (float*)take((size_t)B_SZ * HID * 4);
  float*    Z3  = (float*)take((size_t)B_SZ * D_OUT * 4);
  char* zero0 = ws + off;                                  // ---- zeroed block ----
  float*    c1  = (float*)take((size_t)B_SZ * HID * 4);
  float*    v1  = (float*)take((size_t)B_SZ * HID * 4);
  _Float16* s1h = (_Float16*)take((size_t)B_SZ * HID * 2);
  float*    c2  = (float*)take((size_t)B_SZ * HID * 4);
  float*    v2  = (float*)take((size_t)B_SZ * HID * 4);
  _Float16* s2h = (_Float16*)take((size_t)B_SZ * HID * 2);
  float*    c3  = (float*)take((size_t)B_SZ * D_OUT * 4);
  float*    v3  = (float*)take((size_t)B_SZ * D_OUT * 4);
  _Float16* s3h = (_Float16*)take((size_t)B_SZ * D_OUT * 2);
  float*    act = (float*)take((size_t)B_SZ * D_OUT * 4);
  size_t zbytes = (size_t)((ws + off) - zero0);
  unsigned* stats = (unsigned*)take((size_t)T_STEPS * 3 * 8 * 4);       // 96 slots x 32B

  // ---- per-call re-init (graph-replay safe) ----
  hipMemsetAsync(zero0, 0, zbytes, stream);
  init_stats<<<3, 256, 0, stream>>>(stats, T_STEPS * 3 * 8);

  // ---- stage weights to f16 ----
  f32_to_f16<<<(HID * D_IN + 255) / 256, 256, 0, stream>>>(W1, W1h, HID * D_IN);
  f32_to_f16<<<(HID * HID + 255) / 256, 256, 0, stream>>>(W2, W2h, HID * HID);
  f32_to_f16<<<(D_OUT * HID + 255) / 256, 256, 0, stream>>>(W3, W3h, D_OUT * HID);

  // ---- transpose/convert spikes: (B, D_IN, T) f32 -> (T, B, D_IN) f16 ----
  transpose_spikes<<<dim3(D_IN / 32, B_SZ), dim3(32, 8), 0, stream>>>(spikes, Xh);

  // ---- hoisted layer-1 GEMM for all timesteps: M = T*B ----
  gemm_f16_wmma<<<dim3(HID / 64, (T_STEPS * B_SZ) / 64), 128, 0, stream>>>(
      Xh, W1h, b1, Z1, T_STEPS * B_SZ, HID, D_IN);

  // ---- sequential recurrence ----
  for (int t = 0; t < T_STEPS; ++t) {
    const int tp = (t > 0) ? (t - 1) : 0;
    unsigned* p1 = stats + ((size_t)tp * 3 + 0) * 8;
    unsigned* p2 = stats + ((size_t)tp * 3 + 1) * 8;
    unsigned* p3 = stats + ((size_t)tp * 3 + 2) * 8;
    unsigned* q1 = stats + ((size_t)t * 3 + 0) * 8;
    unsigned* q2 = stats + ((size_t)t * 3 + 1) * 8;
    unsigned* q3 = stats + ((size_t)t * 3 + 2) * 8;

    lif_step<<<(B_SZ * HID) / 256, 256, 0, stream>>>(
        Z1 + (size_t)t * B_SZ * HID, c1, v1, s1h, p1, q1, nullptr, B_SZ * HID, t);

    gemm_f16_wmma<<<dim3(HID / 64, B_SZ / 64), 128, 0, stream>>>(
        s1h, W2h, b2, Z2, B_SZ, HID, HID);

    lif_step<<<(B_SZ * HID) / 256, 256, 0, stream>>>(
        Z2, c2, v2, s2h, p2, q2, nullptr, B_SZ * HID, t);

    gemm_f16_wmma<<<dim3((D_OUT + 63) / 64, B_SZ / 64), 128, 0, stream>>>(
        s2h, W3h, b3, Z3, B_SZ, D_OUT, HID);

    lif_step<<<(B_SZ * D_OUT) / 256, 256, 0, stream>>>(
        Z3, c3, v3, s3h, p3, q3, act, B_SZ * D_OUT, t);
  }

  finalize_out<<<(B_SZ * D_OUT) / 256, 256, 0, stream>>>(act, (float*)d_out, B_SZ * D_OUT);
}
